// SparseTransformer_87531433492810
// MI455X (gfx1250) — matrix operand order repaired and checked by reading, not compiled, not run on hardware
//
#include <hip/hip_runtime.h>
#include <hip/hip_bf16.h>

#define B   128
#define S   52
#define D   1024
#define H   16
#define L   8
#define DFF 4096
#define DK  64
#define BS  (B * S)
#define EPS 1e-5f
#define ATTN_SCALE 0.125f  // 1/sqrt(64)

typedef __bf16 bf16;
typedef __attribute__((ext_vector_type(16))) __bf16 v16bf;
typedef __attribute__((ext_vector_type(8)))  __bf16 v8bf;
typedef __attribute__((ext_vector_type(8)))  float  v8f;

// ---------------------------------------------------------------------------
// fp32 -> bf16 elementwise conversion (weights + activation mirrors)
// ---------------------------------------------------------------------------
__global__ void f2bf_kernel(const float* __restrict__ in, bf16* __restrict__ out, long n) {
    long i = (long)blockIdx.x * blockDim.x + threadIdx.x;
    if (i < n) out[i] = (bf16)in[i];
}

// ---------------------------------------------------------------------------
// token-scatter embedding: sep positions -> sep_token, else action_emb[x[...,2]]
// ---------------------------------------------------------------------------
__global__ void embed_kernel(const int* __restrict__ x, const float* __restrict__ emb,
                             const float* __restrict__ sep_tok, float* __restrict__ h) {
    int bs = blockIdx.x;                 // 0..BS-1
    const int* t = x + (long)bs * 3;
    int t0 = t[0], t1 = t[1], t2 = t[2];
    bool sep = (t0 == 100) && (t1 == 0) && (t2 == 2);
    const float* src = sep ? sep_tok : (emb + (long)t2 * D);
    float* dst = h + (long)bs * D;
    for (int d = threadIdx.x; d < D; d += blockDim.x) dst[d] = src[d];
}

// ---------------------------------------------------------------------------
// BatchNorm1d(S): per-seq-position stats over (B, D)
// ---------------------------------------------------------------------------
__global__ void bn_stats_kernel(const float* __restrict__ h, float* __restrict__ mu,
                                float* __restrict__ var) {
    int s = blockIdx.x;
    float sum = 0.f, sq = 0.f;
    for (int i = threadIdx.x; i < B * D; i += blockDim.x) {
        int b = i >> 10;          // D = 1024
        int d = i & 1023;
        float v = h[((long)b * S + s) * D + d];
        sum += v; sq += v * v;
    }
    __shared__ float ssum[256], ssq[256];
    ssum[threadIdx.x] = sum; ssq[threadIdx.x] = sq;
    __syncthreads();
    for (int st = 128; st > 0; st >>= 1) {
        if (threadIdx.x < st) {
            ssum[threadIdx.x] += ssum[threadIdx.x + st];
            ssq[threadIdx.x]  += ssq[threadIdx.x + st];
        }
        __syncthreads();
    }
    if (threadIdx.x == 0) {
        float m = ssum[0] / (float)(B * D);
        mu[s]  = m;
        var[s] = ssq[0] / (float)(B * D) - m * m;
    }
}

__global__ void bn_apply_kernel(float* __restrict__ h, bf16* __restrict__ hb,
                                const float* __restrict__ mu, const float* __restrict__ var,
                                const float* __restrict__ g, const float* __restrict__ beta) {
    long i = (long)blockIdx.x * blockDim.x + threadIdx.x;
    if (i >= (long)BS * D) return;
    int s = (int)((i / D) % S);
    float val = (h[i] - mu[s]) * rsqrtf(var[s] + EPS) * g[s] + beta[s];
    h[i]  = val;
    hb[i] = (bf16)val;
}

// ---------------------------------------------------------------------------
// bf16 WMMA GEMM: out[M,N] = A[M,K] @ W[N,K]^T + bias  (optional relu)
// Block tile 128x128, 4 waves (2x2), 64x64 per wave = 4x4 v_wmma tiles.
// A/B tiles double-buffered in LDS via GLOBAL_LOAD_ASYNC_TO_LDS_B128
// (ASYNCcnt), K-chunk = 64. Fragments read from LDS with ds_load_b128
// using the CDNA5 wave32 WMMA lane layouts.
// ---------------------------------------------------------------------------
#define KC 64
__global__ void __launch_bounds__(128)
gemm_bf16_kernel(const bf16* __restrict__ A, const bf16* __restrict__ W,
                 const float* __restrict__ bias, float* __restrict__ outF,
                 bf16* __restrict__ outB, int M, int N, int K, int relu) {
    __shared__ bf16 Abuf[2][128 * KC];   // 16 KB x2
    __shared__ bf16 Bbuf[2][128 * KC];   // 16 KB x2

    const int tid   = threadIdx.x;
    const int lane  = tid & 31;
    const int wave  = tid >> 5;        // 0..3
    const int waveM = wave >> 1;       // 0..1
    const int waveN = wave & 1;        // 0..1
    const int mrow  = lane & 15;
    const int half  = lane >> 4;       // 0/1 lane-half of the wave
    const long mblock = (long)blockIdx.y * 128;
    const long nblock = (long)blockIdx.x * 128;
    const int  mwoff  = waveM * 64;
    const int  nwoff  = waveN * 64;

    // async-stage one 128x64 A tile + 128x64 B tile into LDS buffer `buf`
    auto stage = [&](int buf, int kk) {
#pragma unroll
        for (int i = 0; i < 8; i++) {
            int c   = tid + i * 128;       // 0..1023 16B-chunks per matrix
            int row = c >> 3;
            int kc  = c & 7;
            const bf16* ga = A + (mblock + row) * (long)K + kk + kc * 8;
            unsigned la = (unsigned)(uintptr_t)&Abuf[buf][row * KC + kc * 8];
            asm volatile("global_load_async_to_lds_b128 %0, %1, off"
                         :: "v"(la), "v"(ga) : "memory");
            const bf16* gb = W + (nblock + row) * (long)K + kk + kc * 8;
            unsigned lb = (unsigned)(uintptr_t)&Bbuf[buf][row * KC + kc * 8];
            asm volatile("global_load_async_to_lds_b128 %0, %1, off"
                         :: "v"(lb), "v"(gb) : "memory");
        }
    };

    v8f acc[4][4] = {};
    union AF { v16bf v; v8bf h[2]; };

    stage(0, 0);
    asm volatile("s_wait_asynccnt 0x0" ::: "memory");
    __syncthreads();

    for (int kk = 0; kk < K; kk += KC) {
        const int cur = (kk / KC) & 1;
        const bool more = (kk + KC) < K;
        if (more) stage(cur ^ 1, kk + KC);

#pragma unroll
        for (int ks = 0; ks < KC; ks += 32) {
            // A-fragment (16x32 bf16): lane<16: K [ks..+7],[ks+16..+23]
            //                          lane>=16: K [ks+8..+15],[ks+24..+31]
            AF a[4];
#pragma unroll
            for (int mi = 0; mi < 4; mi++) {
                const bf16* ar = &Abuf[cur][(mwoff + mi * 16 + mrow) * KC + ks];
                a[mi].h[0] = *(const v8bf*)(ar + half * 8);
                a[mi].h[1] = *(const v8bf*)(ar + 16 + half * 8);
            }
            // B-fragment (32x16): col n = lane&15 (= W row); lane<16: K [ks..+7],[ks+16..+23]
            // lane>=16: K [ks+8..+15],[ks+24..+31] (same order as A)
            v16bf bf[4];
#pragma unroll
            for (int ni = 0; ni < 4; ni++) {
                const bf16* br = &Bbuf[cur][(nwoff + ni * 16 + mrow) * KC + ks];
                AF t;
                t.h[0] = *(const v8bf*)(br + half * 8);
                t.h[1] = *(const v8bf*)(br + 16 + half * 8);
                bf[ni] = t.v;
            }
#pragma unroll
            for (int mi = 0; mi < 4; mi++)
#pragma unroll
                for (int ni = 0; ni < 4; ni++)
                    acc[mi][ni] = __builtin_amdgcn_wmma_f32_16x16x32_bf16(
                        false, a[mi].v, false, bf[ni], (short)0, acc[mi][ni], false, false);
        }

        if (more) {
            asm volatile("s_wait_asynccnt 0x0" ::: "memory");
        }
        __syncthreads();
    }

    // C/D layout: VGPR r, lane -> m = r + 8*(lane>>4), n = lane&15
#pragma unroll
    for (int mi = 0; mi < 4; mi++) {
#pragma unroll
        for (int ni = 0; ni < 4; ni++) {
            long n   = nblock + nwoff + ni * 16 + mrow;
            float bv = bias[n];
#pragma unroll
            for (int r = 0; r < 8; r++) {
                long m    = mblock + mwoff + mi * 16 + r + half * 8;
                float val = acc[mi][ni][r] + bv;
                if (relu) val = val > 0.f ? val : 0.f;
                long idx = m * (long)N + n;
                if (outF) outF[idx] = val;
                if (outB) outB[idx] = (bf16)val;
            }
        }
    }
}

// ---------------------------------------------------------------------------
// Attention per (b, head): scores, exact top-5 threshold, s*(s>=thr),
// softmax, attn @ V. Tiny vs the GEMMs -> fp32 VALU in LDS.
// ---------------------------------------------------------------------------
__global__ void __launch_bounds__(128)
attention_kernel(const float* __restrict__ q, const float* __restrict__ k,
                 const float* __restrict__ v, const int* __restrict__ x,
                 bf16* __restrict__ ob) {
    __shared__ float qs[S][DK], ks[S][DK], vs[S][DK];
    __shared__ float sc[S][S];
    __shared__ float keyBias[S];
    const int bh  = blockIdx.x;
    const int b   = bh / H;
    const int hd  = bh % H;
    const int tid = threadIdx.x;
    const long base = (long)b * S * D + hd * DK;

    for (int i = tid; i < S * DK; i += blockDim.x) {
        int s = i / DK, d = i % DK;
        long g = base + (long)s * D + d;
        qs[s][d] = q[g]; ks[s][d] = k[g]; vs[s][d] = v[g];
    }
    if (tid < S) {
        const int* t = x + ((long)b * S + tid) * 3;
        bool pad = (t[0] == 100) && (t[1] == 0) && (t[2] == 0);
        keyBias[tid] = pad ? -1e9f : 0.f;
    }
    __syncthreads();

    for (int idx = tid; idx < S * S; idx += blockDim.x) {
        int i = idx / S, j = idx % S;
        float acc = 0.f;
#pragma unroll
        for (int d = 0; d < DK; d++) acc += qs[i][d] * ks[j][d];
        sc[i][j] = (keyBias[j] < 0.f) ? -1e9f : acc * ATTN_SCALE;
    }
    __syncthreads();

    if (tid < S) {
        const int i = tid;
        // exact top-5 via register insertion
        float t0 = -3e38f, t1 = -3e38f, t2 = -3e38f, t3 = -3e38f, t4 = -3e38f;
        for (int j = 0; j < S; j++) {
            float vj = sc[i][j];
            if (vj > t4) {
                if (vj > t0)      { t4 = t3; t3 = t2; t2 = t1; t1 = t0; t0 = vj; }
                else if (vj > t1) { t4 = t3; t3 = t2; t2 = t1; t1 = vj; }
                else if (vj > t2) { t4 = t3; t3 = t2; t2 = vj; }
                else if (vj > t3) { t4 = t3; t3 = vj; }
                else              { t4 = vj; }
            }
        }
        float thr = t4;
        float mx = -3e38f;
        for (int j = 0; j < S; j++) {
            float sj = sc[i][j];
            float sp = (sj >= thr) ? sj : 0.f;   // scores * (scores >= thr)
            sc[i][j] = sp;
            mx = sp > mx ? sp : mx;
        }
        float sum = 0.f;
        for (int j = 0; j < S; j++) {
            float e = __expf(sc[i][j] - mx);
            sc[i][j] = e; sum += e;
        }
        float inv = 1.f / sum;
        for (int j = 0; j < S; j++) sc[i][j] *= inv;
    }
    __syncthreads();

    for (int idx = tid; idx < S * DK; idx += blockDim.x) {
        int i = idx / DK, d = idx % DK;
        float acc = 0.f;
        for (int j = 0; j < S; j++) acc += sc[i][j] * vs[j][d];
        ob[base + (long)i * D + d] = (bf16)acc;
    }
}

// ---------------------------------------------------------------------------
// h = LayerNorm(h + y) * g + b ; writes fp32 h and bf16 mirror
// ---------------------------------------------------------------------------
__global__ void __launch_bounds__(256)
add_ln_kernel(float* __restrict__ h, const float* __restrict__ y,
              const float* __restrict__ g, const float* __restrict__ beta,
              bf16* __restrict__ hb) {
    const long row = blockIdx.x;
    float* hr = h + row * D;
    const float* yr = y + row * D;
    float vals[4];
    float sum = 0.f;
#pragma unroll
    for (int i = 0; i < 4; i++) {
        int d = threadIdx.x + i * 256;
        vals[i] = hr[d] + yr[d];
        sum += vals[i];
    }
    __shared__ float red[256];
    red[threadIdx.x] = sum; __syncthreads();
    for (int st = 128; st > 0; st >>= 1) {
        if (threadIdx.x < st) red[threadIdx.x] += red[threadIdx.x + st];
        __syncthreads();
    }
    float mean = red[0] / (float)D;
    __syncthreads();
    float sq = 0.f;
#pragma unroll
    for (int i = 0; i < 4; i++) { float dlt = vals[i] - mean; sq += dlt * dlt; }
    red[threadIdx.x] = sq; __syncthreads();
    for (int st = 128; st > 0; st >>= 1) {
        if (threadIdx.x < st) red[threadIdx.x] += red[threadIdx.x + st];
        __syncthreads();
    }
    float inv = rsqrtf(red[0] / (float)D + EPS);
#pragma unroll
    for (int i = 0; i < 4; i++) {
        int d = threadIdx.x + i * 256;
        float o = (vals[i] - mean) * inv * g[d] + beta[d];
        hr[d] = o;
        hb[row * D + d] = (bf16)o;
    }
}

// ---------------------------------------------------------------------------
// out[b, :] = h[b, S-1, :] @ Wout^T + bout
// ---------------------------------------------------------------------------
__global__ void final_kernel(const float* __restrict__ h, const float* __restrict__ Wout,
                             const float* __restrict__ bout, float* __restrict__ out) {
    int i = blockIdx.x * blockDim.x + threadIdx.x;
    if (i >= B * 3) return;
    int b = i / 3, c = i % 3;
    const float* hr = h + ((long)b * S + (S - 1)) * D;
    const float* wr = Wout + (long)c * D;
    float acc = bout[c];
    for (int d = 0; d < D; d++) acc += hr[d] * wr[d];
    out[i] = acc;
}

// ---------------------------------------------------------------------------
extern "C" void kernel_launch(void* const* d_in, const int* in_sizes, int n_in,
                              void* d_out, int out_size, void* d_ws, size_t ws_size,
                              hipStream_t stream) {
    (void)in_sizes; (void)n_in; (void)out_size; (void)ws_size;
    const int*   x       = (const int*)  d_in[0];
    const float* emb     = (const float*)d_in[1];
    const float* sep_tok = (const float*)d_in[2];
    const float* bn_g    = (const float*)d_in[3];
    const float* bn_b    = (const float*)d_in[4];
    const float* Wq      = (const float*)d_in[5];
    const float* bq      = (const float*)d_in[6];
    const float* Wk      = (const float*)d_in[7];
    const float* bk      = (const float*)d_in[8];
    const float* Wv      = (const float*)d_in[9];
    const float* bv      = (const float*)d_in[10];
    const float* Wo      = (const float*)d_in[11];
    const float* bo      = (const float*)d_in[12];
    const float* W1      = (const float*)d_in[13];
    const float* b1      = (const float*)d_in[14];
    const float* W2      = (const float*)d_in[15];
    const float* b2      = (const float*)d_in[16];
    const float* ln1g    = (const float*)d_in[17];
    const float* ln1b    = (const float*)d_in[18];
    const float* ln2g    = (const float*)d_in[19];
    const float* ln2b    = (const float*)d_in[20];
    const float* Wout    = (const float*)d_in[21];
    const float* bout    = (const float*)d_in[22];
    float* out = (float*)d_out;

    char* ws = (char*)d_ws;
    size_t off = 0;
    auto alloc = [&](size_t bytes) -> void* {
        void* p = ws + off;
        off += (bytes + 255) & ~(size_t)255;
        return p;
    };
    float* h   = (float*)alloc((size_t)BS * D * 4);
    bf16*  hb  = (bf16*) alloc((size_t)BS * D * 2);
    float* qb  = (float*)alloc((size_t)BS * D * 4);
    float* kb  = (float*)alloc((size_t)BS * D * 4);
    float* vb  = (float*)alloc((size_t)BS * D * 4);
    bf16*  ob  = (bf16*) alloc((size_t)BS * D * 2);
    float* yb  = (float*)alloc((size_t)BS * D * 4);
    bf16*  fb  = (bf16*) alloc((size_t)BS * DFF * 2);
    float* mu  = (float*)alloc(S * 4);
    float* var = (float*)alloc(S * 4);
    bf16* wqb = (bf16*)alloc((size_t)L * D * D * 2);
    bf16* wkb = (bf16*)alloc((size_t)L * D * D * 2);
    bf16* wvb = (bf16*)alloc((size_t)L * D * D * 2);
    bf16* wob = (bf16*)alloc((size_t)L * D * D * 2);
    bf16* w1b = (bf16*)alloc((size_t)L * DFF * D * 2);
    bf16* w2b = (bf16*)alloc((size_t)L * D * DFF * 2);

    auto cvt = [&](const float* src, bf16* dst, long n) {
        unsigned blocks = (unsigned)((n + 255) / 256);
        f2bf_kernel<<<blocks, 256, 0, stream>>>(src, dst, n);
    };
    cvt(Wq, wqb, (long)L * D * D);
    cvt(Wk, wkb, (long)L * D * D);
    cvt(Wv, wvb, (long)L * D * D);
    cvt(Wo, wob, (long)L * D * D);
    cvt(W1, w1b, (long)L * DFF * D);
    cvt(W2, w2b, (long)L * D * DFF);

    embed_kernel<<<BS, 256, 0, stream>>>(x, emb, sep_tok, h);
    bn_stats_kernel<<<S, 256, 0, stream>>>(h, mu, var);
    bn_apply_kernel<<<(BS * D) / 256, 256, 0, stream>>>(h, hb, mu, var, bn_g, bn_b);

    dim3 gD(D / 128, BS / 128);       // 8 x 52
    dim3 gF(DFF / 128, BS / 128);     // 32 x 52
    for (int l = 0; l < L; l++) {
        size_t wOff  = (size_t)l * D * D;
        size_t w1Off = (size_t)l * DFF * D;
        size_t w2Off = (size_t)l * D * DFF;
        gemm_bf16_kernel<<<gD, 128, 0, stream>>>(hb, wqb + wOff, bq + l * D, qb, nullptr, BS, D, D, 0);
        gemm_bf16_kernel<<<gD, 128, 0, stream>>>(hb, wkb + wOff, bk + l * D, kb, nullptr, BS, D, D, 0);
        gemm_bf16_kernel<<<gD, 128, 0, stream>>>(hb, wvb + wOff, bv + l * D, vb, nullptr, BS, D, D, 0);
        attention_kernel<<<B * H, 128, 0, stream>>>(qb, kb, vb, x, ob);
        gemm_bf16_kernel<<<gD, 128, 0, stream>>>(ob, wob + wOff, bo + l * D, yb, nullptr, BS, D, D, 0);
        add_ln_kernel<<<BS, 256, 0, stream>>>(h, yb, ln1g + l * D, ln1b + l * D, hb);
        gemm_bf16_kernel<<<gF, 128, 0, stream>>>(hb, w1b + w1Off, b1 + l * DFF, nullptr, fb, BS, DFF, D, 1);
        gemm_bf16_kernel<<<gD, 128, 0, stream>>>(fb, w2b + w2Off, b2 + l * D, yb, nullptr, BS, D, DFF, 0);
        add_ln_kernel<<<BS, 256, 0, stream>>>(h, yb, ln2g + l * D, ln2b + l * D, hb);
    }
    final_kernel<<<(B * 3 + 127) / 128, 128, 0, stream>>>(h, Wout, bout, out);
}
